// LowLevelPolicyNetworknoskill_30958124270358
// MI455X (gfx1250) — compile-verified, matrix operand order checked by reading, not run on hardware
//
#include <hip/hip_runtime.h>

// ---------------------------------------------------------------------------
// Types for CDNA5 WMMA (wave32)
// ---------------------------------------------------------------------------
typedef __attribute__((ext_vector_type(16))) __bf16 v16bf;
typedef __attribute__((ext_vector_type(8)))  float  v8f;
typedef unsigned short ushort_t;

__device__ __forceinline__ unsigned short f2bf(float f) {
  union { float f; unsigned u; } c; c.f = f;
  unsigned u = c.u;
  unsigned r = u + 0x7FFFu + ((u >> 16) & 1u);   // round-to-nearest-even
  return (unsigned short)(r >> 16);
}
__device__ __forceinline__ float bf2f(unsigned short h) {
  union { unsigned u; float f; } c; c.u = ((unsigned)h) << 16;
  return c.f;
}

// ---------------------------------------------------------------------------
// One-shot f32 -> bf16 conversion (weights + network inputs). n % 4 == 0.
// ---------------------------------------------------------------------------
__global__ __launch_bounds__(256)
void f32_to_bf16_kernel(const float* __restrict__ src,
                        ushort_t* __restrict__ dst, long long n)
{
  long long i = ((long long)blockIdx.x * 256 + threadIdx.x) * 4;
  if (i >= n) return;
  float4 v = *(const float4*)(src + i);
  uint2 p;
  p.x = ((unsigned)f2bf(v.y) << 16) | f2bf(v.x);
  p.y = ((unsigned)f2bf(v.w) << 16) | f2bf(v.z);
  *(uint2*)(dst + i) = p;
}

// ---------------------------------------------------------------------------
// GEMM:  C[M,N] = A[M,K](bf16) * B[N,K](bf16)^T + bias[N](f32)
//   aRowMap: 0 = identity, 1 = gather over padded layout (r -> r/512*513+r%512)
//   cRowMap: 0 = identity, 1 = scatter into padded layout
//   outBf16: 1 = store bf16, 0 = store f32
// Block tile 128x128, K-step 32, double-buffered LDS, 8 waves in 2(M)x4(N),
// each wave 64x32 = 4x2 WMMAs per K-step. Staging = pure bf16 copy:
// one uint2 global load + cndmask + one ds_store_b64 per 4-element unit.
// ---------------------------------------------------------------------------
#define BM 128
#define BN 128
#define BK 32
#define LDT 48             // padded LDS row stride in ushorts (96B, 16B aligned)
#define TILEU (BM * LDT)

__device__ __forceinline__ void stage_rows_bf(const ushort_t* __restrict__ src,
                                              int ld, int rowMap, int row0,
                                              int nrows, int k0, int K, int tid,
                                              uint2* st)
{
#pragma unroll
  for (int e = 0; e < 4; ++e) {
    int u = tid + e * 256;              // 0..1023 : unit index
    int r = u >> 3;                     // tile row 0..127
    int c = (u & 7) * 4;                // tile col 0,4,..,28
    int gr = row0 + r;
    int grc = gr < nrows ? gr : nrows - 1;               // clamp (branchless)
    long long ar = rowMap ? (long long)((grc >> 9) * 513 + (grc & 511))
                          : (long long)grc;
    int kc  = k0 + c;
    int kcc = (kc + 4 <= K) ? kc : (K - 4);              // clamp (K % 4 == 0)
    uint2 v = *(const uint2*)(src + ar * ld + kcc);      // unconditional, 8B ok
    bool ok = (gr < nrows) && (kc < K);
    st[e].x = ok ? v.x : 0u;
    st[e].y = ok ? v.y : 0u;
  }
}

__device__ __forceinline__ void store_tile(ushort_t* dst, int tid,
                                           const uint2* st)
{
#pragma unroll
  for (int e = 0; e < 4; ++e) {
    int u = tid + e * 256;
    int r = u >> 3;
    int c = (u & 7) * 4;
    *(uint2*)&dst[r * LDT + c] = st[e];
  }
}

__global__ __launch_bounds__(256)
void gemm_bf16_wmma(const ushort_t* __restrict__ A, int lda, int aRowMap,
                    const ushort_t* __restrict__ B,
                    const float* __restrict__ bias,
                    void* __restrict__ Cv, int ldc, int cRowMap, int cColOff,
                    int M, int N, int K, int relu, int outBf16)
{
  __shared__ __align__(16) ushort_t As[2 * TILEU];
  __shared__ __align__(16) ushort_t Bs[2 * TILEU];

  const int tid  = threadIdx.x;
  const int lane = tid & 31;
  const int wave = tid >> 5;
  const int wm   = wave >> 2;        // 0..1  (M sub-block of 64)
  const int wn   = wave & 3;         // 0..3  (N sub-block of 32)
  const int tileM = blockIdx.y * BM;
  const int tileN = blockIdx.x * BN;

  v8f acc[4][2];
#pragma unroll
  for (int im = 0; im < 4; ++im)
#pragma unroll
    for (int in = 0; in < 2; ++in)
#pragma unroll
      for (int r = 0; r < 8; ++r) acc[im][in][r] = 0.0f;

  const int mrow = lane & 15;
  const int kb   = (lane >> 4) << 3;   // 0 or 8 (bf16 elements)

  uint2 sa[4], sb[4];
  const int ktiles = (K + BK - 1) / BK;

  // Prologue: stage tile 0 into buffer 0.
  stage_rows_bf(A, lda, aRowMap, tileM, M, 0, K, tid, sa);
  stage_rows_bf(B, K,   0,       tileN, N, 0, K, tid, sb);
  store_tile(As, tid, sa);
  store_tile(Bs, tid, sb);
  __syncthreads();

  for (int kt = 0; kt < ktiles; ++kt) {
    const ushort_t* curA = As + (kt & 1) * TILEU;
    const ushort_t* curB = Bs + (kt & 1) * TILEU;
    const bool more = (kt + 1) < ktiles;

    // Issue next tile's global loads early (overlap with WMMA below).
    if (more) {
      stage_rows_bf(A, lda, aRowMap, tileM, M, (kt + 1) * BK, K, tid, sa);
      stage_rows_bf(B, K,   0,       tileN, N, (kt + 1) * BK, K, tid, sb);
    }

    // Fragment loads straight from LDS (wave32 layout) + 8 WMMAs.
    union Frag { v16bf v; uint4 q[2]; };
    Frag af[4], bfr[2];
#pragma unroll
    for (int im = 0; im < 4; ++im) {
      int m = wm * 64 + im * 16 + mrow;
      af[im].q[0] = *(const uint4*)&curA[m * LDT + kb];
      af[im].q[1] = *(const uint4*)&curA[m * LDT + kb + 16];
    }
#pragma unroll
    for (int in = 0; in < 2; ++in) {
      int n = wn * 32 + in * 16 + mrow;
      bfr[in].q[0] = *(const uint4*)&curB[n * LDT + kb];
      bfr[in].q[1] = *(const uint4*)&curB[n * LDT + kb + 16];
    }
#pragma unroll
    for (int im = 0; im < 4; ++im)
#pragma unroll
      for (int in = 0; in < 2; ++in)
        acc[im][in] = __builtin_amdgcn_wmma_f32_16x16x32_bf16(
            false, af[im].v, false, bfr[in].v,
            (short)0, acc[im][in], false, false);

    // Write next tile into the other buffer while this one is being consumed.
    if (more) {
      store_tile(As + ((kt + 1) & 1) * TILEU, tid, sa);
      store_tile(Bs + ((kt + 1) & 1) * TILEU, tid, sb);
    }
    __syncthreads();
  }

  // ---- epilogue: bias (+ReLU) + row/col remap store ----
  const int mhalf = (lane >> 4) * 8;
#pragma unroll
  for (int im = 0; im < 4; ++im) {
#pragma unroll
    for (int in = 0; in < 2; ++in) {
      int n = tileN + wn * 32 + in * 16 + (lane & 15);
      if (n >= N) continue;
      float bv = bias ? bias[n] : 0.0f;
#pragma unroll
      for (int r = 0; r < 8; ++r) {
        int m = tileM + wm * 64 + im * 16 + mhalf + r;
        if (m >= M) continue;
        float v = acc[im][in][r] + bv;
        if (relu) v = fmaxf(v, 0.0f);
        int cr = (cRowMap == 1) ? ((m >> 9) * 513 + (m & 511)) : m;
        long long idx = (long long)cr * ldc + cColOff + n;
        if (outBf16) ((ushort_t*)Cv)[idx] = f2bf(v);
        else         ((float*)Cv)[idx]    = v;
      }
    }
  }
}

// ---------------------------------------------------------------------------
// Banded attention: window of 17 keys (i-16..i), + pad additive mask.
// One 64-thread block per (b, h, 64-query tile). K/V window staged in LDS (f32).
// QKV is bf16; softmax math in f32; output written bf16.
// ---------------------------------------------------------------------------
#define AQB 64
__global__ __launch_bounds__(64)
void attn_band_kernel(const ushort_t* __restrict__ QKV,  // (16*513, 1536) bf16
                      const float* __restrict__ padadd,  // (16*513) f32
                      ushort_t* __restrict__ Oout,       // (16*513, 512) bf16
                      int n)
{
  const int b  = blockIdx.x;
  const int h  = blockIdx.y;
  const int q0 = blockIdx.z * AQB;
  const int tid = threadIdx.x;

  __shared__ float Ks[(AQB + 16) * 64];
  __shared__ float Vs[(AQB + 16) * 64];

  int kstart = q0 - 16; if (kstart < 0) kstart = 0;
  int kend   = q0 + AQB; if (kend > n) kend = n;
  const int rows = kend - kstart;
  const long long rowbase = (long long)b * 513;

  for (int i = tid; i < rows * 64; i += 64) {
    int r = i >> 6, d = i & 63;
    long long row = rowbase + kstart + r;
    Ks[i] = bf2f(QKV[row * 1536 + 512  + h * 64 + d]);
    Vs[i] = bf2f(QKV[row * 1536 + 1024 + h * 64 + d]);
  }
  __syncthreads();

  const int q = q0 + tid;
  if (q >= n) return;

  float qv[64];
  const long long qrow = rowbase + q;
#pragma unroll
  for (int d = 0; d < 64; ++d) qv[d] = bf2f(QKV[qrow * 1536 + h * 64 + d]);

  float sc[17];
  float mx = -1e30f;
#pragma unroll
  for (int jj = 0; jj < 17; ++jj) {
    int j = q - 16 + jj;
    float s = -1e30f;
    if (j >= 0) {
      const float* kp = &Ks[(j - kstart) * 64];
      float a = 0.0f;
#pragma unroll
      for (int d = 0; d < 64; ++d) a += qv[d] * kp[d];
      s = a * 0.125f + padadd[b * 513 + j];
    }
    sc[jj] = s;
    mx = fmaxf(mx, s);
  }
  float sum = 0.0f;
#pragma unroll
  for (int jj = 0; jj < 17; ++jj) { sc[jj] = expf(sc[jj] - mx); sum += sc[jj]; }
  const float inv = 1.0f / sum;

  for (int d = 0; d < 64; ++d) {
    float o = 0.0f;
#pragma unroll
    for (int jj = 0; jj < 17; ++jj) {
      int j = q - 16 + jj;
      if (j >= 0) o += sc[jj] * Vs[(j - kstart) * 64 + d];
    }
    Oout[qrow * 512 + h * 64 + d] = f2bf(o * inv);
  }
}

// ---------------------------------------------------------------------------
// Fused residual + LayerNorm over 512-wide bf16 rows.  X = LN(X + R) * g + bt
// Stats and normalization in f32; result stored bf16.
// ---------------------------------------------------------------------------
__global__ __launch_bounds__(256)
void resid_ln_kernel(ushort_t* __restrict__ X, const ushort_t* __restrict__ R,
                     const float* __restrict__ g, const float* __restrict__ bt)
{
  const long long row = blockIdx.x;
  const int tid = threadIdx.x;
  __shared__ float s_sum[256];
  __shared__ float s_sq[256];

  float v0 = bf2f(X[row * 512 + tid])       + bf2f(R[row * 512 + tid]);
  float v1 = bf2f(X[row * 512 + tid + 256]) + bf2f(R[row * 512 + tid + 256]);
  s_sum[tid] = v0 + v1;
  s_sq[tid]  = v0 * v0 + v1 * v1;
  __syncthreads();
  for (int off = 128; off > 0; off >>= 1) {
    if (tid < off) { s_sum[tid] += s_sum[tid + off]; s_sq[tid] += s_sq[tid + off]; }
    __syncthreads();
  }
  const float mean = s_sum[0] * (1.0f / 512.0f);
  const float var  = s_sq[0]  * (1.0f / 512.0f) - mean * mean;
  const float rstd = rsqrtf(var + 1e-5f);
  X[row * 512 + tid]       = f2bf((v0 - mean) * rstd * g[tid]       + bt[tid]);
  X[row * 512 + tid + 256] = f2bf((v1 - mean) * rstd * g[tid + 256] + bt[tid + 256]);
}

// ---------------------------------------------------------------------------
// Pad mask: padadd[b,j] = -1e9 if goal[b,j,:] == -1 everywhere (j<512), else 0
// (reads the original f32 goal input)
// ---------------------------------------------------------------------------
__global__ void pad_mask_kernel(const float* __restrict__ goal,
                                float* __restrict__ padadd)
{
  int idx = blockIdx.x * 256 + threadIdx.x;      // b*513 + j
  if (idx >= 16 * 513) return;
  int b = idx / 513, j = idx % 513;
  float add = 0.0f;
  if (j < 512) {
    const float* gp = goal + ((long long)b * 512 + j) * 300;
    bool all = true;
    for (int t = 0; t < 300; ++t)
      if (gp[t] != -1.0f) { all = false; break; }
    if (all) add = -1e9f;
  }
  padadd[idx] = add;
}

// Zero the appended token row (pos 512 of each batch) in bf16 X.
__global__ void zero_pad_rows_kernel(ushort_t* __restrict__ X)
{
  int idx = blockIdx.x * 256 + threadIdx.x;      // 16*512
  if (idx >= 16 * 512) return;
  int b = idx >> 9, c = idx & 511;
  X[((long long)b * 513 + 512) * 512 + c] = 0;
}

// ---------------------------------------------------------------------------
// Orchestration
// ---------------------------------------------------------------------------
static inline void launch_gemm(const ushort_t* A, int lda, int aMap,
                               const ushort_t* B, const float* bias,
                               void* C, int ldc, int cMap, int cOff,
                               int M, int N, int K, int relu, int outBf16,
                               hipStream_t s)
{
  dim3 grid((N + BN - 1) / BN, (M + BM - 1) / BM);
  gemm_bf16_wmma<<<grid, 256, 0, s>>>(A, lda, aMap, B, bias, C, ldc, cMap, cOff,
                                      M, N, K, relu, outBf16);
}

static inline void cvt(const float* src, ushort_t* dst, long long n,
                       hipStream_t s)
{
  long long blocks = (n / 4 + 255) / 256;
  f32_to_bf16_kernel<<<(unsigned)blocks, 256, 0, s>>>(src, dst, n);
}

extern "C" void kernel_launch(void* const* d_in, const int* in_sizes, int n_in,
                              void* d_out, int out_size, void* d_ws, size_t ws_size,
                              hipStream_t stream)
{
  const float* state  = (const float*)d_in[0];
  const float* goal   = (const float*)d_in[1];
  const float* W_obs  = (const float*)d_in[2];
  const float* b_obs  = (const float*)d_in[3];
  const float* W_lang = (const float*)d_in[4];
  const float* b_lang = (const float*)d_in[5];
  const float* W_in   = (const float*)d_in[6];
  const float* b_in   = (const float*)d_in[7];
  const float* Wqkv   = (const float*)d_in[8];
  const float* bqkv   = (const float*)d_in[9];
  const float* Wo     = (const float*)d_in[10];
  const float* bo     = (const float*)d_in[11];
  const float* W1     = (const float*)d_in[12];
  const float* b1     = (const float*)d_in[13];
  const float* W2     = (const float*)d_in[14];
  const float* b2     = (const float*)d_in[15];
  const float* g1     = (const float*)d_in[16];
  const float* bt1    = (const float*)d_in[17];
  const float* g2     = (const float*)d_in[18];
  const float* bt2    = (const float*)d_in[19];
  const float* W_outp = (const float*)d_in[20];
  const float* b_outp = (const float*)d_in[21];
  const float* W_a1   = (const float*)d_in[22];
  const float* b_a1   = (const float*)d_in[23];
  const float* W_a2   = (const float*)d_in[24];
  const float* b_a2   = (const float*)d_in[25];
  float* out = (float*)d_out;

  const int NTOK = 16 * 512;       // 8192
  const int NPAD = 16 * 513;       // 8208
  const int n    = 513;

  // ---- Workspace carve-up (bf16 activations + bf16 weight copies) ----
  char* w = (char*)d_ws;
  auto allocBF = [&](long long cnt) { ushort_t* p = (ushort_t*)w; w += cnt * 2; return p; };
  ushort_t* Xb    = allocBF((long long)NPAD * 512);   // residual stream
  ushort_t* BIGb  = allocBF((long long)NPAD * 2048);  // QKV / FF1
  ushort_t* T1b   = allocBF((long long)NPAD * 512);   // U / attn-out / FF2
  ushort_t* T2b   = allocBF((long long)NPAD * 512);   // O / H
  ushort_t* stb   = allocBF((long long)NTOK * 768);   // state (bf16)
  ushort_t* glb   = allocBF((long long)NTOK * 300);   // goal  (bf16)
  ushort_t* Wobsb  = allocBF(256 * 768);
  ushort_t* Wlangb = allocBF(256 * 300);
  ushort_t* Winb   = allocBF(512 * 512);
  ushort_t* Wqkvb  = allocBF(3LL * 1536 * 512);
  ushort_t* Wob    = allocBF(3LL * 512 * 512);
  ushort_t* W1b    = allocBF(3LL * 2048 * 512);
  ushort_t* W2b    = allocBF(3LL * 512 * 2048);
  ushort_t* Woutpb = allocBF(256 * 512);
  ushort_t* Wa1b   = allocBF(12 * 256);
  ushort_t* Wa2b   = allocBF(89 * 256);
  float*    PAD    = (float*)w;  w += (long long)NPAD * 4;
  (void)ws_size; (void)in_sizes; (void)n_in; (void)out_size;

  // ---- One-shot conversions (inputs + weights) ----
  cvt(state,  stb,    (long long)NTOK * 768, stream);
  cvt(goal,   glb,    (long long)NTOK * 300, stream);
  cvt(W_obs,  Wobsb,  256 * 768,        stream);
  cvt(W_lang, Wlangb, 256 * 300,        stream);
  cvt(W_in,   Winb,   512 * 512,        stream);
  cvt(Wqkv,   Wqkvb,  3LL * 1536 * 512, stream);
  cvt(Wo,     Wob,    3LL * 512 * 512,  stream);
  cvt(W1,     W1b,    3LL * 2048 * 512, stream);
  cvt(W2,     W2b,    3LL * 512 * 2048, stream);
  cvt(W_outp, Woutpb, 256 * 512,        stream);
  cvt(W_a1,   Wa1b,   12 * 256,         stream);
  cvt(W_a2,   Wa2b,   89 * 256,         stream);

  // Pad mask
  pad_mask_kernel<<<(NPAD + 255) / 256, 256, 0, stream>>>(goal, PAD);

  // ---- Embed: st | lg -> U (T1b), then X = U @ W_in^T + b_in (padded rows) ----
  launch_gemm(stb, 768, 0, Wobsb,  b_obs,  T1b, 512, 0, 0,   NTOK, 256, 768, 0, 1, stream);
  launch_gemm(glb, 300, 0, Wlangb, b_lang, T1b, 512, 0, 256, NTOK, 256, 300, 0, 1, stream);
  launch_gemm(T1b, 512, 0, Winb,   b_in,   Xb,  512, 1, 0,   NTOK, 512, 512, 0, 1, stream);
  zero_pad_rows_kernel<<<(NTOK + 255) / 256, 256, 0, stream>>>(Xb);

  // ---- Encoder layers ----
  for (int l = 0; l < 3; ++l) {
    const ushort_t* Wqkv_l = Wqkvb + (long long)l * 1536 * 512;
    const float*    bqkv_l = bqkv  + (long long)l * 1536;
    const ushort_t* Wo_l   = Wob   + (long long)l * 512 * 512;
    const float*    bo_l   = bo    + (long long)l * 512;
    const ushort_t* W1_l   = W1b   + (long long)l * 2048 * 512;
    const float*    b1_l   = b1    + (long long)l * 2048;
    const ushort_t* W2_l   = W2b   + (long long)l * 512 * 2048;
    const float*    b2_l   = b2    + (long long)l * 512;

    // QKV
    launch_gemm(Xb, 512, 0, Wqkv_l, bqkv_l, BIGb, 1536, 0, 0, NPAD, 1536, 512, 0, 1, stream);
    // Banded attention -> T1b
    dim3 agrid(16, 8, (n + AQB - 1) / AQB);
    attn_band_kernel<<<agrid, AQB, 0, stream>>>(BIGb, PAD, T1b, n);
    // Output projection -> T2b
    launch_gemm(T1b, 512, 0, Wo_l, bo_l, T2b, 512, 0, 0, NPAD, 512, 512, 0, 1, stream);
    // X = LN(X + T2b)
    resid_ln_kernel<<<NPAD, 256, 0, stream>>>(Xb, T2b, g1 + l * 512, bt1 + l * 512);
    // FF1 (ReLU) -> BIGb
    launch_gemm(Xb, 512, 0, W1_l, b1_l, BIGb, 2048, 0, 0, NPAD, 2048, 512, 1, 1, stream);
    // FF2 -> T1b
    launch_gemm(BIGb, 2048, 0, W2_l, b2_l, T1b, 512, 0, 0, NPAD, 512, 2048, 0, 1, stream);
    // X = LN(X + T1b)
    resid_ln_kernel<<<NPAD, 256, 0, stream>>>(Xb, T1b, g2 + l * 512, bt2 + l * 512);
  }

  // ---- Output heads: H = X[:, :512] @ W_outp^T ; out = [H@W_a1^T | H@W_a2^T] ----
  launch_gemm(Xb,  512, 1, Woutpb, b_outp, T2b, 256, 0, 0,  NTOK, 256, 512, 0, 1, stream);
  launch_gemm(T2b, 256, 0, Wa1b,   b_a1,   out, 101, 0, 0,  NTOK, 12,  256, 0, 0, stream);
  launch_gemm(T2b, 256, 0, Wa2b,   b_a2,   out, 101, 0, 12, NTOK, 89,  256, 0, 0, stream);
}
